// GAT_12687333392403
// MI455X (gfx1250) — matrix operand order repaired and checked by reading, not compiled, not run on hardware
//
#include <hip/hip_runtime.h>
#include <hip/hip_bf16.h>

typedef __attribute__((ext_vector_type(16))) _Float16 v16h;
typedef __attribute__((ext_vector_type(8)))  _Float16 v8h;
typedef __attribute__((ext_vector_type(8)))  float    v8f;

#define NEG_SLOPE 0.2f

// ---------- helpers: ordered-int encoding for float atomic max ----------
__device__ __forceinline__ int f32_to_ordered(float f) {
    int i = __float_as_int(f);
    return (i >= 0) ? i : (i ^ 0x7fffffff);
}
__device__ __forceinline__ float ordered_to_f32(int i) {
    return __int_as_float((i >= 0) ? i : (i ^ 0x7fffffff));
}

// ---------- small utility kernels ----------
__global__ void fill_i32_kernel(int* p, int v, int n) {
    int i = blockIdx.x * blockDim.x + threadIdx.x;
    if (i < n) p[i] = v;
}

__global__ void f32_to_f16_kernel(const float* __restrict__ in, _Float16* __restrict__ out, int n) {
    int i = blockIdx.x * blockDim.x + threadIdx.x;
    if (i < n) out[i] = (_Float16)in[i];
}

// Wt[n][k] = W[k][n]  (transpose + convert to f16 so B fragments are contiguous)
__global__ void transpose_w_f16_kernel(const float* __restrict__ W, _Float16* __restrict__ Wt,
                                       int K, int Nout) {
    int i = blockIdx.x * blockDim.x + threadIdx.x;
    if (i >= K * Nout) return;
    int k = i / Nout, n = i % Nout;
    Wt[(size_t)n * K + k] = (_Float16)W[i];
}

// ---------- register-blocked WMMA GEMM ----------
// C[M,NOUT] = A[M,K](f16 rowmajor) x Bt[NOUT,K](f16 rowmajor, B pre-transposed).
// One wave computes a 16 x (16*NB) strip of C: A fragment loaded once per k-step,
// reused across NB independent WMMAs (8x less A traffic, 8 WMMAs per load-wait).
// K/NOUT compile-time -> fully unrolled, stores use immediate offsets.
template <int K, int NOUT, int NB>
__global__ void gemm_wmma_f16(const _Float16* __restrict__ A,
                              const _Float16* __restrict__ Bt,
                              float* __restrict__ C,
                              int Mtiles) {
    constexpr int NT = NOUT / 16;   // total N tiles
    constexpr int NG = NT / NB;     // N groups (waves per tile-row)
    static_assert(NT % NB == 0, "NB must divide NOUT/16");

    int wave = blockIdx.x * (blockDim.x >> 5) + (threadIdx.x >> 5);
    if (wave >= Mtiles * NG) return;   // wave-uniform exit: EXEC stays all-ones for WMMA
    const int lane = threadIdx.x & 31;
    const int tm = wave / NG;
    const int tg = wave % NG;
    const int row0 = tm << 4;
    const int col0 = tg * (NB * 16);
    const int m16 = lane & 15;         // A row / B col / C col within tile
    const int g   = lane >> 4;         // lane group

    const _Float16* arow  = A  + (size_t)(row0 + m16) * K;
    const _Float16* bbase = Bt + (size_t)(col0 + m16) * K;   // tile j is +j*16 columns -> +j*16*K halves

    v8f acc[NB];
#pragma unroll
    for (int j = 0; j < NB; ++j) acc[j] = (v8f){};

#pragma unroll
    for (int k0 = 0; k0 < K; k0 += 32) {
        // A 16x32 f16 fragment: lane (m,g) holds K = {g*8..g*8+7} U {16+g*8..16+g*8+7}
        v8h alo = *(const v8h*)(arow + k0 + g * 8);
        v8h ahi = *(const v8h*)(arow + k0 + 16 + g * 8);
        v16h a = __builtin_shufflevector(alo, ahi, 0,1,2,3,4,5,6,7,8,9,10,11,12,13,14,15);
#pragma unroll
        for (int j = 0; j < NB; ++j) {
            // B 32x16 f16 fragment: lane (n,g) holds column n, K = {g*8..g*8+7} U {16+g*8..16+g*8+7}
            v8h blo = *(const v8h*)(bbase + (size_t)j * 16 * K + k0 + g * 8);
            v8h bhi = *(const v8h*)(bbase + (size_t)j * 16 * K + k0 + 16 + g * 8);
            v16h b = __builtin_shufflevector(blo, bhi, 0,1,2,3,4,5,6,7,8,9,10,11,12,13,14,15);
            acc[j] = __builtin_amdgcn_wmma_f32_16x16x32_f16(false, a, false, b,
                                                            (short)0, acc[j], false, false);
        }
    }

    // C/D layout: VGPR v, lane (n,g) -> element (M = v + 8*g, N = n); NOUT is compile-time
    float* cbase = C + (size_t)(row0 + 8 * g) * NOUT + col0 + m16;
#pragma unroll
    for (int j = 0; j < NB; ++j) {
#pragma unroll
        for (int v = 0; v < 8; ++v) {
            cbase[v * NOUT + j * 16] = acc[j][v];
        }
    }
}

// ---------- edge pipeline (dim fixed = 32, heads parameterized) ----------
__device__ __forceinline__ void edge_endpoints(const long long* ei, int E_, int e,
                                               long long& s, long long& d) {
    if (e < E_) { s = ei[e]; d = ei[(size_t)E_ + e]; }
    else        { s = d = (long long)(e - E_); }
}

// pass 1: score[e][h] = sum_d att[h][d] * leaky_relu(xl[s][h][d] + xr[d][h][d]); atomic-max into smax[d][h]
__global__ void edge_score_kernel(const long long* __restrict__ ei, int E_, int N_,
                                  const float* __restrict__ xl, const float* __restrict__ xr,
                                  const float* __restrict__ att,
                                  float* __restrict__ score, int* __restrict__ smax, int heads) {
    int idx = blockIdx.x * blockDim.x + threadIdx.x;
    if (idx >= (E_ + N_) * heads) return;
    int e = idx / heads, h = idx % heads;
    long long s, d;
    edge_endpoints(ei, E_, e, s, d);
    const float* pl = xl + ((size_t)s * heads + h) * 32;
    const float* pr = xr + ((size_t)d * heads + h) * 32;
    const float* pa = att + h * 32;
    float acc = 0.f;
#pragma unroll
    for (int i = 0; i < 32; ++i) {
        float v = pl[i] + pr[i];
        v = (v > 0.f) ? v : NEG_SLOPE * v;
        acc += pa[i] * v;
    }
    score[idx] = acc;
    atomicMax(&smax[(size_t)d * heads + h], f32_to_ordered(acc));
}

// pass 2: score <- exp(score - smax[dst]); atomic-add into denom[dst]
__global__ void edge_exp_kernel(const long long* __restrict__ ei, int E_, int N_,
                                float* __restrict__ score, const int* __restrict__ smax,
                                float* __restrict__ denom, int heads) {
    int idx = blockIdx.x * blockDim.x + threadIdx.x;
    if (idx >= (E_ + N_) * heads) return;
    int e = idx / heads, h = idx % heads;
    long long s, d;
    edge_endpoints(ei, E_, e, s, d);
    float m = ordered_to_f32(smax[(size_t)d * heads + h]);
    float ex = expf(score[idx] - m);
    score[idx] = ex;
    atomicAdd(&denom[(size_t)d * heads + h], ex);
}

// pass 3: agg[dst][h][:] += xl[src][h][:] * (ex / (denom[dst][h] + 1e-16))
__global__ void edge_aggregate_kernel(const long long* __restrict__ ei, int E_, int N_,
                                      const float* __restrict__ score, const float* __restrict__ denom,
                                      const float* __restrict__ xl, float* __restrict__ agg, int heads) {
    int idx = blockIdx.x * blockDim.x + threadIdx.x;
    if (idx >= (E_ + N_) * heads) return;
    int e = idx / heads, h = idx % heads;
    long long s, d;
    edge_endpoints(ei, E_, e, s, d);
    float alpha = score[idx] / (denom[(size_t)d * heads + h] + 1e-16f);
    const float4* pl = (const float4*)(xl + ((size_t)s * heads + h) * 32);
    float* po = agg + ((size_t)d * heads + h) * 32;
#pragma unroll
    for (int i = 0; i < 8; ++i) {
        float4 v = pl[i];
        atomicAdd(po + i * 4 + 0, v.x * alpha);
        atomicAdd(po + i * 4 + 1, v.y * alpha);
        atomicAdd(po + i * 4 + 2, v.z * alpha);
        atomicAdd(po + i * 4 + 3, v.w * alpha);
    }
}

// layer-1 epilogue: h = elu(agg + b1), convert to f16 for layer-2 GEMMs
__global__ void finalize1_kernel(const float* __restrict__ agg, const float* __restrict__ b1,
                                 _Float16* __restrict__ h1h, int total /* N*256 */) {
    int i = blockIdx.x * blockDim.x + threadIdx.x;
    if (i >= total) return;
    float v = agg[i] + b1[i & 255];
    v = (v > 0.f) ? v : (expf(v) - 1.f);
    h1h[i] = (_Float16)v;
}

// layer-2 epilogue + linear heads: h = agg2 + b2; out1 = h@Wh1+bh1; out2 = h@Wh2+bh2
__global__ void finalize2_kernel(const float* __restrict__ agg2, const float* __restrict__ b2,
                                 const float* __restrict__ Wh1, const float* __restrict__ bh1,
                                 const float* __restrict__ Wh2, const float* __restrict__ bh2,
                                 float* __restrict__ out, int N_) {
    int n = blockIdx.x * blockDim.x + threadIdx.x;
    if (n >= N_) return;
    float h[32];
#pragma unroll
    for (int i = 0; i < 32; ++i) h[i] = agg2[(size_t)n * 32 + i] + b2[i];

    float* o1 = out + (size_t)n * 4;
    float* o2 = out + (size_t)N_ * 4 + (size_t)n * 3;
    float* oh = out + (size_t)N_ * 7 + (size_t)n * 32;
#pragma unroll
    for (int j = 0; j < 4; ++j) {
        float acc = bh1[j];
#pragma unroll
        for (int i = 0; i < 32; ++i) acc += h[i] * Wh1[i * 4 + j];
        o1[j] = acc;
    }
#pragma unroll
    for (int j = 0; j < 3; ++j) {
        float acc = bh2[j];
#pragma unroll
        for (int i = 0; i < 32; ++i) acc += h[i] * Wh2[i * 3 + j];
        o2[j] = acc;
    }
#pragma unroll
    for (int i = 0; i < 32; ++i) oh[i] = h[i];
}

// ---------- host side ----------
static inline int blocks_for(long long n, int bs) { return (int)((n + bs - 1) / bs); }

extern "C" void kernel_launch(void* const* d_in, const int* in_sizes, int n_in,
                              void* d_out, int out_size, void* d_ws, size_t ws_size,
                              hipStream_t stream) {
    const float*     x    = (const float*)d_in[0];
    const long long* ei   = (const long long*)d_in[1];   // int64 edge_index [2,E]
    const float*     Wl1  = (const float*)d_in[2];
    const float*     Wr1  = (const float*)d_in[3];
    const float*     att1 = (const float*)d_in[4];
    const float*     b1   = (const float*)d_in[5];
    const float*     Wl2  = (const float*)d_in[6];
    const float*     Wr2  = (const float*)d_in[7];
    const float*     att2 = (const float*)d_in[8];
    const float*     b2   = (const float*)d_in[9];
    const float*     Wh1  = (const float*)d_in[10];
    const float*     bh1  = (const float*)d_in[11];
    const float*     Wh2  = (const float*)d_in[12];
    const float*     bh2  = (const float*)d_in[13];
    float* out = (float*)d_out;

    const int N_ = in_sizes[0] / 128;      // 100000 (divisible by 16)
    const int E_ = in_sizes[1] / 2;        // 1600000
    const int EN = E_ + N_;

    // ---- workspace layout (phase-reused, 256B aligned) ----
    char* ws = (char*)d_ws;
    size_t off = 0;
    auto carve = [&](size_t bytes) { void* p = ws + off; off += (bytes + 255) & ~(size_t)255; return p; };

    _Float16* xh    = (_Float16*)carve((size_t)N_ * 256 * sizeof(_Float16)); // phase A: xh [N,128]; phase B: h1h [N,256]
    _Float16* h1h   = xh;
    _Float16* Wl1t  = (_Float16*)carve((size_t)256 * 128 * sizeof(_Float16));
    _Float16* Wr1t  = (_Float16*)carve((size_t)256 * 128 * sizeof(_Float16));
    _Float16* Wl2t  = (_Float16*)carve((size_t)32 * 256 * sizeof(_Float16));
    _Float16* Wr2t  = (_Float16*)carve((size_t)32 * 256 * sizeof(_Float16));
    float*    xl    = (float*)carve((size_t)N_ * 256 * sizeof(float));       // xl1 [N,256]; later xl2 [N,32]
    float*    xr    = (float*)carve((size_t)N_ * 256 * sizeof(float));       // xr1 [N,256]; later xr2 [N,32]
    float*    score = (float*)carve((size_t)EN * 8 * sizeof(float));         // layer1 [EN,8]; layer2 [EN,1]
    int*      smax  = (int*)carve((size_t)N_ * 8 * sizeof(int));
    float*    denom = (float*)carve((size_t)N_ * 8 * sizeof(float));
    float*    agg   = (float*)carve((size_t)N_ * 256 * sizeof(float));       // agg1 [N,256]; later agg2 [N,32]

    const int BS = 256;

    // ---- phase A: precision convert + weight transpose ----
    f32_to_f16_kernel<<<blocks_for((long long)N_ * 128, BS), BS, 0, stream>>>(x, xh, N_ * 128);
    transpose_w_f16_kernel<<<blocks_for(128 * 256, BS), BS, 0, stream>>>(Wl1, Wl1t, 128, 256);
    transpose_w_f16_kernel<<<blocks_for(128 * 256, BS), BS, 0, stream>>>(Wr1, Wr1t, 128, 256);
    transpose_w_f16_kernel<<<blocks_for(256 * 32, BS), BS, 0, stream>>>(Wl2, Wl2t, 256, 32);
    transpose_w_f16_kernel<<<blocks_for(256 * 32, BS), BS, 0, stream>>>(Wr2, Wr2t, 256, 32);

    // ---- layer-1 GEMMs: [N,128] x [128,256] via WMMA; one wave per 16x128 strip (NB=8) ----
    {
        int Mt = N_ >> 4;
        int waves = Mt * (256 / 16 / 8);   // Mt * NG, NG = 2
        gemm_wmma_f16<128, 256, 8><<<blocks_for(waves, 8), 256, 0, stream>>>(xh, Wl1t, xl, Mt);
        gemm_wmma_f16<128, 256, 8><<<blocks_for(waves, 8), 256, 0, stream>>>(xh, Wr1t, xr, Mt);
    }

    // ---- layer-1 edge softmax + aggregation ----
    fill_i32_kernel<<<blocks_for((long long)N_ * 8, BS), BS, 0, stream>>>(smax, (int)0x80000000, N_ * 8);
    hipMemsetAsync(denom, 0, (size_t)N_ * 8 * sizeof(float), stream);
    hipMemsetAsync(agg, 0, (size_t)N_ * 256 * sizeof(float), stream);
    {
        long long tot = (long long)EN * 8;
        edge_score_kernel<<<blocks_for(tot, BS), BS, 0, stream>>>(ei, E_, N_, xl, xr, att1, score, smax, 8);
        edge_exp_kernel<<<blocks_for(tot, BS), BS, 0, stream>>>(ei, E_, N_, score, smax, denom, 8);
        edge_aggregate_kernel<<<blocks_for(tot, BS), BS, 0, stream>>>(ei, E_, N_, score, denom, xl, agg, 8);
    }
    finalize1_kernel<<<blocks_for((long long)N_ * 256, BS), BS, 0, stream>>>(agg, b1, h1h, N_ * 256);

    // ---- layer-2 GEMMs: [N,256] x [256,32] via WMMA; one wave per 16x32 strip (NB=2) ----
    {
        int Mt = N_ >> 4;
        int waves = Mt;                    // NG = 1
        gemm_wmma_f16<256, 32, 2><<<blocks_for(waves, 8), 256, 0, stream>>>(h1h, Wl2t, xl, Mt);
        gemm_wmma_f16<256, 32, 2><<<blocks_for(waves, 8), 256, 0, stream>>>(h1h, Wr2t, xr, Mt);
    }

    // ---- layer-2 edge softmax + aggregation (heads = 1) ----
    fill_i32_kernel<<<blocks_for((long long)N_, BS), BS, 0, stream>>>(smax, (int)0x80000000, N_);
    hipMemsetAsync(denom, 0, (size_t)N_ * sizeof(float), stream);
    hipMemsetAsync(agg, 0, (size_t)N_ * 32 * sizeof(float), stream);
    {
        long long tot = (long long)EN;
        edge_score_kernel<<<blocks_for(tot, BS), BS, 0, stream>>>(ei, E_, N_, xl, xr, att2, score, smax, 1);
        edge_exp_kernel<<<blocks_for(tot, BS), BS, 0, stream>>>(ei, E_, N_, score, smax, denom, 1);
        edge_aggregate_kernel<<<blocks_for(tot, BS), BS, 0, stream>>>(ei, E_, N_, score, denom, xl, agg, 1);
    }

    // ---- epilogue: h = agg + b2; out1/out2 heads; write (out1 | out2 | h) into d_out ----
    finalize2_kernel<<<blocks_for((long long)N_, BS), BS, 0, stream>>>(agg, b2, Wh1, bh1, Wh2, bh2, out, N_);
}
